// MOGConv_35794257444923
// MI455X (gfx1250) — matrix operand order repaired and checked by reading, not compiled, not run on hardware
//
#include <hip/hip_runtime.h>
#include <hip/hip_bf16.h>

typedef __attribute__((ext_vector_type(16))) _Float16 v16h;
typedef __attribute__((ext_vector_type(8)))  _Float16 v8h;
typedef __attribute__((ext_vector_type(8)))  float    v8f;
typedef __attribute__((ext_vector_type(4)))  int      v4i;

#define NPTS   8192
#define GRP    1024
#define HCH    64
#define KNBR   20
#define KNNTOT 360
#define LCOLS  (NPTS * KNBR)   // 163840

// ---------------------------------------------------------------------------
// CDNA5 async global->LDS staging (ASYNCcnt tracked). Builtin signature per
// hipcc diagnostic: (int4 addrspace(1)*, int4 addrspace(3)*, imm offset, imm cpol).
// Guarded so the file still compiles (synchronous 16B copy) if absent;
// host pass always takes the fallback.
// ---------------------------------------------------------------------------
#if defined(__AMDGCN__) && \
    __has_builtin(__builtin_amdgcn_global_load_async_to_lds_b128) && \
    __has_builtin(__builtin_amdgcn_s_wait_asynccnt)
#define USE_ASYNC_LDS 1
#endif

__device__ __forceinline__ void async_cp16(const _Float16* g, _Float16* l) {
#ifdef USE_ASYNC_LDS
  __builtin_amdgcn_global_load_async_to_lds_b128(
      (__attribute__((address_space(1))) v4i*)g,
      (__attribute__((address_space(3))) v4i*)l, 0, 0);
#else
  *(v8h*)l = *(const v8h*)g;
#endif
}
__device__ __forceinline__ void async_wait_all() {
#ifdef USE_ASYNC_LDS
  __builtin_amdgcn_s_wait_asynccnt(0);
#endif
}

// ---------------------------------------------------------------------------
// WMMA fragment helpers (ISA 7.12.2 layouts, wave32)
//   A 16x32 f16: a[h] = A[lane%16][ (h<8?0:16) + (lane/16)*8 + (h&7) ]
//   B 32x16 f16 staged transposed (BsT[col][k]): b[h] = BsT[lane%16][ (h<8?0:16) + (lane/16)*8 + (h&7) ]
//   C/D 16x16 f32: c[v] = C[v + (lane/16)*8][lane%16]
// ---------------------------------------------------------------------------
__device__ __forceinline__ v16h load_afrag(const _Float16* rowp) {
  v16h a;
#pragma unroll
  for (int i = 0; i < 8; ++i) { a[i] = rowp[i]; a[i + 8] = rowp[16 + i]; }
  return a;
}
__device__ __forceinline__ v16h load_bfrag(const _Float16* colp) {
  v16h b;
#pragma unroll
  for (int i = 0; i < 16; ++i) b[i] = colp[i];
  return b;
}
__device__ __forceinline__ float lrelu(float v) { return v >= 0.f ? v : 0.2f * v; }

// ---------------------------------------------------------------------------
// kNN: one block per query point; candidates = its contiguous 1024-pt group.
// Bitonic sort (dist,idx) in LDS; keep first 360 ascending.
// ---------------------------------------------------------------------------
__global__ void knn_kernel(const float* __restrict__ x, int* __restrict__ idxb) {
  __shared__ float pts[GRP * 3];
  __shared__ float dist[GRP];
  __shared__ int   lidx[GRP];
  const int q = blockIdx.x;
  const int base = (q / GRP) * GRP;
  const int tid = threadIdx.x;

  for (int i = tid; i < GRP * 3; i += 256) pts[i] = x[base * 3 + i];
  __syncthreads();

  const int ql = q - base;
  const float qx = pts[ql * 3 + 0], qy = pts[ql * 3 + 1], qz = pts[ql * 3 + 2];
  for (int j = tid; j < GRP; j += 256) {
    float dx = pts[j * 3 + 0] - qx, dy = pts[j * 3 + 1] - qy, dz = pts[j * 3 + 2] - qz;
    float d = dx * dx + dy * dy + dz * dz;
    dist[j] = (j == ql) ? 3.4e38f : d;
    lidx[j] = base + j;
  }
  __syncthreads();

  for (int ksz = 2; ksz <= GRP; ksz <<= 1) {
    for (int j = ksz >> 1; j > 0; j >>= 1) {
      for (int i = tid; i < GRP; i += 256) {
        int ixj = i ^ j;
        if (ixj > i) {
          bool up = ((i & ksz) == 0);
          float di = dist[i], dj = dist[ixj];
          if ((di > dj) == up) {
            dist[i] = dj; dist[ixj] = di;
            int t = lidx[i]; lidx[i] = lidx[ixj]; lidx[ixj] = t;
          }
        }
      }
      __syncthreads();
    }
  }
  for (int t = tid; t < KNNTOT; t += 256) idxb[q * KNNTOT + t] = lidx[t];
}

// ---------------------------------------------------------------------------
// Stage-1 edge GEMM (C=3, f32 coords): Y[64, N*20] = W1[64,6] @ f
// ---------------------------------------------------------------------------
__global__ void gemm_edge3_kernel(const float* __restrict__ W,
                                  const float* __restrict__ xt,   // [N,3]
                                  const int* __restrict__ idxb,
                                  float* __restrict__ Y, int ncols) {
  constexpr int C = 3, KP = 32, M = 64, MT = 4, COLS = 32, LD = KP + 8;
  __shared__ _Float16 As[M * LD];
  __shared__ _Float16 BsT[COLS * LD];
  const int tid = threadIdx.x;

  for (int i = tid; i < M * KP; i += 256) {
    int m = i / KP, k = i % KP;
    As[m * LD + k] = (k < 2 * C) ? (_Float16)W[m * (2 * C) + k] : (_Float16)0.f;
  }
  const int colbase = blockIdx.x * COLS;
  {
    const int c = tid & 31;
    const int col = colbase + c;
    const int n = col / KNBR, kk = col % KNBR;
    const int nb = idxb[n * KNNTOT + kk];
    for (int k = tid >> 5; k < KP; k += 8) {
      float v;
      if (k < C)          v = xt[nb * C + k] - xt[n * C + k];
      else if (k < 2 * C) v = xt[n * C + (k - C)];
      else                v = 0.f;
      BsT[c * LD + k] = (_Float16)v;
    }
  }
  __syncthreads();

  const int wid = tid >> 5, lane = tid & 31, lo = lane & 15, hi = lane >> 4;
  const int mt = wid % MT, ct = wid / MT;
  v8f acc = {};
#pragma unroll
  for (int kb = 0; kb < KP; kb += 32) {
    v16h a = load_afrag(&As[(mt * 16 + lo) * LD + kb + hi * 8]);
    v16h b = load_afrag(&BsT[(ct * 16 + lo) * LD + kb + hi * 8]);
    acc = __builtin_amdgcn_wmma_f32_16x16x32_f16(false, a, false, b, (short)0, acc, false, false);
  }
  const int col = colbase + ct * 16 + lo;
#pragma unroll
  for (int v = 0; v < 8; ++v)
    Y[(size_t)(mt * 16 + hi * 8 + v) * ncols + col] = acc[v];
}

// ---------------------------------------------------------------------------
// Edge GEMM stages 2-4 (C=64, f16 [N,64] activations):
//   center half (ctr) of the edge feature is a raw copy -> async b128 to LDS;
//   diff half (nb-ctr) computed on VALU.
// ---------------------------------------------------------------------------
__global__ void gemm_edge16_kernel(const float* __restrict__ W,        // [64,128]
                                   const _Float16* __restrict__ xt16,  // [N,64]
                                   const int* __restrict__ idxb, int stride,
                                   float* __restrict__ Y, int ncols) {
  constexpr int KP = 128, M = 64, MT = 4, COLS = 32, LD = KP + 8;
  __shared__ _Float16 As[M * LD];
  __shared__ _Float16 BsT[COLS * LD];
  const int tid = threadIdx.x;

  for (int i = tid; i < M * KP; i += 256) {
    int m = i / KP, k = i % KP;
    As[m * LD + k] = (_Float16)W[m * KP + k];
  }
  const int colbase = blockIdx.x * COLS;
  const int c = tid & 31;
  const int col = colbase + c;
  const int n = col / KNBR, kk = col % KNBR;
  const int nb = idxb[n * KNNTOT + kk * stride];

  {  // async copy of the center half: BsT[c][64+k] = xt16[n*64+k], 64 f16 = 8x16B
    const int t = tid >> 5;  // 0..7
    async_cp16(xt16 + (size_t)n * 64 + t * 8, &BsT[c * LD + 64 + t * 8]);
  }
  for (int k = tid >> 5; k < 64; k += 8)  // diff half on VALU
    BsT[c * LD + k] =
        (_Float16)((float)xt16[(size_t)nb * 64 + k] - (float)xt16[(size_t)n * 64 + k]);
  async_wait_all();
  __syncthreads();

  const int wid = tid >> 5, lane = tid & 31, lo = lane & 15, hi = lane >> 4;
  const int mt = wid % MT, ct = wid / MT;
  v8f acc = {};
#pragma unroll
  for (int kb = 0; kb < KP; kb += 32) {
    v16h a = load_afrag(&As[(mt * 16 + lo) * LD + kb + hi * 8]);
    v16h b = load_afrag(&BsT[(ct * 16 + lo) * LD + kb + hi * 8]);
    acc = __builtin_amdgcn_wmma_f32_16x16x32_f16(false, a, false, b, (short)0, acc, false, false);
  }
  const int ocol = colbase + ct * 16 + lo;
#pragma unroll
  for (int v = 0; v < 8; ++v)
    Y[(size_t)(mt * 16 + hi * 8 + v) * ncols + ocol] = acc[v];
}

// ---------------------------------------------------------------------------
// Norm-fused GEMM: Y2[64,L] = W[64,64] @ lrelu((Y1-mean)*rstd)  (VALU staging)
// ---------------------------------------------------------------------------
__global__ void gemm_normed_kernel(const float* __restrict__ W,
                                   const float* __restrict__ Yin,
                                   const float* __restrict__ st,
                                   float* __restrict__ Y, int ncols) {
  constexpr int M = 64, KP = 64, MT = 4, COLS = 32, LD = KP + 8;
  __shared__ _Float16 As[M * LD];
  __shared__ _Float16 BsT[COLS * LD];
  const int tid = threadIdx.x;

  for (int i = tid; i < M * KP; i += 256) {
    int m = i / KP, k = i % KP;
    As[m * LD + k] = (_Float16)W[m * KP + k];
  }
  const int colbase = blockIdx.x * COLS;
  for (int i = tid; i < COLS * KP; i += 256) {
    int c = i % COLS, k = i / COLS;
    int col = colbase + c;
    float v = (Yin[(size_t)k * ncols + col] - st[2 * k]) * st[2 * k + 1];
    BsT[c * LD + k] = (_Float16)lrelu(v);
  }
  __syncthreads();

  const int wid = tid >> 5, lane = tid & 31, lo = lane & 15, hi = lane >> 4;
  const int mt = wid % MT, ct = wid / MT;
  v8f acc = {};
#pragma unroll
  for (int kb = 0; kb < KP; kb += 32) {
    v16h a = load_afrag(&As[(mt * 16 + lo) * LD + kb + hi * 8]);
    v16h b = load_afrag(&BsT[(ct * 16 + lo) * LD + kb + hi * 8]);
    acc = __builtin_amdgcn_wmma_f32_16x16x32_f16(false, a, false, b, (short)0, acc, false, false);
  }
  const int col = colbase + ct * 16 + lo;
#pragma unroll
  for (int v = 0; v < 8; ++v)
    Y[(size_t)(mt * 16 + hi * 8 + v) * ncols + col] = acc[v];
}

// ---------------------------------------------------------------------------
// Decoder / global-feature GEMM over up to 4 concatenated f16 [N,C] sources.
// B staging = contiguous async b128 runs per column. Optional per-row bias.
// ---------------------------------------------------------------------------
struct Cat4 { const _Float16* p[4]; int r[4]; };

template <int M, int KP>
__global__ void gemm_cat_kernel(const float* __restrict__ A, int lda,
                                Cat4 cat,
                                const float* __restrict__ bias,
                                float* __restrict__ Y, int ncols) {
  constexpr int MT = M / 16;
  constexpr int CT = 8 / MT;
  constexpr int COLS = CT * 16;
  constexpr int KC = (KP > 128) ? KP / 2 : KP;
  constexpr int LD = KC + 8;
  __shared__ _Float16 As[M * LD];
  __shared__ _Float16 BsT[COLS * LD];
  const int tid = threadIdx.x;
  const int colbase = blockIdx.x * COLS;
  const int wid = tid >> 5, lane = tid & 31, lo = lane & 15, hi = lane >> 4;
  const int mt = wid % MT, ct = wid / MT;
  v8f acc = {};

  for (int kc0 = 0; kc0 < KP; kc0 += KC) {
    if (kc0) __syncthreads();
    for (int i = tid; i < M * KC; i += 256) {
      int m = i / KC, kk = i % KC;
      As[m * LD + kk] = (_Float16)A[m * lda + kc0 + kk];
    }
#ifdef USE_ASYNC_LDS
    {
      int base = 0;
      for (int s = 0; s < 4; ++s) {
        const int rs = cat.r[s];
        const int klo = kc0 > base ? kc0 : base;
        const int khi = (kc0 + KC) < (base + rs) ? (kc0 + KC) : (base + rs);
        if (rs > 0 && klo < khi) {
          const int n16 = (khi - klo) >> 3;  // 16B transfers per column
          for (int i = tid; i < COLS * n16; i += 256) {
            int c = i % COLS, t = i / COLS;
            const _Float16* g =
                cat.p[s] + (size_t)(colbase + c) * rs + (klo - base) + t * 8;
            async_cp16(g, &BsT[c * LD + (klo - kc0) + t * 8]);
          }
        }
        base += rs;
      }
      async_wait_all();
    }
#else
    for (int i = tid; i < COLS * KC; i += 256) {
      int c = i % COLS, kk = i / COLS;
      int col = colbase + c;
      int krem = kc0 + kk;
      float v = 0.f;
      for (int s = 0; s < 4; ++s) {
        if (krem >= 0 && krem < cat.r[s])
          v = (float)cat.p[s][(size_t)col * cat.r[s] + krem];
        krem -= cat.r[s];
      }
      BsT[c * LD + kk] = (_Float16)v;
    }
#endif
    __syncthreads();
#pragma unroll
    for (int kb = 0; kb < KC; kb += 32) {
      v16h a = load_afrag(&As[(mt * 16 + lo) * LD + kb + hi * 8]);
      v16h b = load_afrag(&BsT[(ct * 16 + lo) * LD + kb + hi * 8]);
      acc = __builtin_amdgcn_wmma_f32_16x16x32_f16(false, a, false, b, (short)0, acc, false, false);
    }
  }
  const int col = colbase + ct * 16 + lo;
#pragma unroll
  for (int v = 0; v < 8; ++v) {
    int mrow = mt * 16 + hi * 8 + v;
    float o = acc[v] + (bias ? bias[mrow] : 0.f);
    Y[(size_t)mrow * ncols + col] = o;
  }
}

// ---------------------------------------------------------------------------
// Per-channel instance-norm stats: st[2r]=mean, st[2r+1]=rsqrt(var+eps)
// ---------------------------------------------------------------------------
__global__ void chan_stats_kernel(const float* __restrict__ Y, int L,
                                  float* __restrict__ st) {
  const int row = blockIdx.x, tid = threadIdx.x;
  const float* p = Y + (size_t)row * L;
  float s = 0.f, s2 = 0.f;
  for (int i = tid; i < L; i += 256) { float v = p[i]; s += v; s2 += v * v; }
  __shared__ float sh[256], sh2[256];
  sh[tid] = s; sh2[tid] = s2; __syncthreads();
  for (int o = 128; o > 0; o >>= 1) {
    if (tid < o) { sh[tid] += sh[tid + o]; sh2[tid] += sh2[tid + o]; }
    __syncthreads();
  }
  if (tid == 0) {
    float m = sh[0] / (float)L;
    float var = sh2[0] / (float)L - m * m;
    st[2 * row] = m;
    st[2 * row + 1] = rsqrtf(var + 1e-5f);
  }
}

// normalize+lrelu+max over k(20)+residual; write f32 [C,N] + f16 [N,C]
__global__ void finalize2d_kernel(const float* __restrict__ Y2,
                                  const float* __restrict__ st,
                                  const float* __restrict__ resCN,
                                  float* __restrict__ outCN,
                                  _Float16* __restrict__ outNC16) {
  int gid = blockIdx.x * 256 + threadIdx.x;
  if (gid >= NPTS * HCH) return;
  int c = gid & 63, n = gid >> 6;
  float m = st[2 * c], rs = st[2 * c + 1];
  const float* p = Y2 + (size_t)c * LCOLS + (size_t)n * KNBR;
  float mx = -3.4e38f;
#pragma unroll
  for (int k = 0; k < KNBR; ++k) mx = fmaxf(mx, lrelu((p[k] - m) * rs));
  float o = mx + (resCN ? resCN[(size_t)c * NPTS + n] : 0.f);
  outCN[(size_t)c * NPTS + n] = o;
  outNC16[(size_t)n * HCH + c] = (_Float16)o;
}

// g[c] = max_n lrelu((Y[c,n]-m)*rs)
__global__ void rowmax_act_kernel(const float* __restrict__ Y,
                                  const float* __restrict__ st,
                                  float* __restrict__ gvec, int L) {
  const int row = blockIdx.x, tid = threadIdx.x;
  float m = st[2 * row], rs = st[2 * row + 1];
  const float* p = Y + (size_t)row * L;
  float mx = -3.4e38f;
  for (int i = tid; i < L; i += 256) mx = fmaxf(mx, lrelu((p[i] - m) * rs));
  __shared__ float sh[256];
  sh[tid] = mx; __syncthreads();
  for (int o = 128; o > 0; o >>= 1) {
    if (tid < o) sh[tid] = fmaxf(sh[tid], sh[tid + o]);
    __syncthreads();
  }
  if (tid == 0) gvec[row] = sh[0];
}

// normalize + lrelu, write f16 [N,M] (column-major for next GEMM)
__global__ void norm_act_kernel(const float* __restrict__ Y,
                                const float* __restrict__ st,
                                _Float16* __restrict__ out16, int M) {
  int gid = blockIdx.x * 256 + threadIdx.x;
  if (gid >= M * NPTS) return;
  int row = gid / NPTS, n = gid % NPTS;
  float v = lrelu((Y[gid] - st[2 * row]) * st[2 * row + 1]);
  out16[(size_t)n * M + row] = (_Float16)v;
}

// b10[m] = sum_c W10[m, c<128] * g[c]   (broadcast-g column collapses to bias)
__global__ void bias10_kernel(const float* __restrict__ W10,
                              const float* __restrict__ gvec,
                              float* __restrict__ b10) {
  int m = blockIdx.x * 128 + threadIdx.x;
  if (m >= 128) return;
  float s = 0.f;
  for (int c = 0; c < 128; ++c) s += W10[m * 192 + c] * gvec[c];
  b10[m] = s;
}

// per-object mean of x1' (segment mean commutes with Wr); d1 is f16 [N,64]
__global__ void obj_mean_kernel(const _Float16* __restrict__ d1,
                                float* __restrict__ xm) {
  int gid = blockIdx.x * 256 + threadIdx.x;
  if (gid >= 8 * HCH) return;
  int obj = gid / HCH, c = gid % HCH;
  const _Float16* p = d1 + (size_t)obj * GRP * HCH + c;
  float s = 0.f;
  for (int i = 0; i < GRP; ++i) s += (float)p[(size_t)i * HCH];
  xm[obj * HCH + c] = s * (1.f / (float)GRP);
}

// out[obj, c] = Wr[c,:] . xm[obj,:]
__global__ void final_out_kernel(const float* __restrict__ Wr,
                                 const float* __restrict__ xm,
                                 float* __restrict__ out) {
  int gid = blockIdx.x * 256 + threadIdx.x;
  if (gid >= 8 * 128) return;
  int obj = gid / 128, c = gid % 128;
  float s = 0.f;
  for (int k = 0; k < HCH; ++k) s += Wr[c * HCH + k] * xm[obj * HCH + k];
  out[obj * 128 + c] = s;
}

// ---------------------------------------------------------------------------
extern "C" void kernel_launch(void* const* d_in, const int* in_sizes, int n_in,
                              void* d_out, int out_size, void* d_ws, size_t ws_size,
                              hipStream_t stream) {
  (void)in_sizes; (void)n_in; (void)out_size; (void)ws_size;
  const float* x   = (const float*)d_in[0];
  const float* W1  = (const float*)d_in[3];
  const float* W2  = (const float*)d_in[4];
  const float* W3  = (const float*)d_in[5];
  const float* W4  = (const float*)d_in[6];
  const float* W5  = (const float*)d_in[7];
  const float* W6  = (const float*)d_in[8];
  const float* W7  = (const float*)d_in[9];
  const float* W8  = (const float*)d_in[10];
  const float* W9  = (const float*)d_in[11];
  const float* W10 = (const float*)d_in[12];
  const float* W11 = (const float*)d_in[13];
  const float* W12 = (const float*)d_in[14];
  const float* W13 = (const float*)d_in[15];
  const float* Wr  = (const float*)d_in[16];

  size_t off = 0;
  auto alloc = [&](size_t bytes) -> void* {
    void* p = (char*)d_ws + off;
    off += (bytes + 255) & ~(size_t)255;
    return p;
  };
  int*      idxb  = (int*)alloc((size_t)NPTS * KNNTOT * 4);
  float*    y1    = (float*)alloc((size_t)HCH * LCOLS * 4);
  float*    y2    = (float*)alloc((size_t)HCH * LCOLS * 4);
  float*    xCN   = (float*)alloc((size_t)4 * HCH * NPTS * 4);  // x1..x4 f32 [C,N]
  _Float16* xNC16 = (_Float16*)alloc((size_t)4 * NPTS * HCH * 2);
  float*    ydec  = (float*)alloc((size_t)128 * NPTS * 4);      // also yg
  _Float16* d4nc  = (_Float16*)alloc((size_t)NPTS * 128 * 2);
  _Float16* d3nc  = (_Float16*)alloc((size_t)NPTS * 128 * 2);
  _Float16* d2nc  = (_Float16*)alloc((size_t)NPTS * HCH * 2);
  _Float16* d1nc  = (_Float16*)alloc((size_t)NPTS * HCH * 2);
  float*    st    = (float*)alloc(2 * 128 * 4);
  float*    st2   = (float*)alloc(2 * 128 * 4);
  float*    gvec  = (float*)alloc(128 * 4);
  float*    b10   = (float*)alloc(128 * 4);
  float*    xm    = (float*)alloc(8 * HCH * 4);

  float* x1CN = xCN;
  float* x2CN = xCN + (size_t)HCH * NPTS;
  float* x3CN = xCN + 2 * (size_t)HCH * NPTS;
  _Float16* x1nc = xNC16;
  _Float16* x2nc = xNC16 + (size_t)NPTS * HCH;
  _Float16* x3nc = xNC16 + 2 * (size_t)NPTS * HCH;
  _Float16* x4nc = xNC16 + 3 * (size_t)NPTS * HCH;

  const int GEB = LCOLS / 32;          // 5120 blocks per edge gemm
  const int ELT = (NPTS * HCH) / 256;  // 2048 blocks for finalize

  knn_kernel<<<NPTS, 256, 0, stream>>>(x, idxb);

  // ---- stage 1 (C=3) ----
  gemm_edge3_kernel<<<GEB, 256, 0, stream>>>(W1, x, idxb, y1, LCOLS);
  chan_stats_kernel<<<HCH, 256, 0, stream>>>(y1, LCOLS, st);
  gemm_normed_kernel<<<GEB, 256, 0, stream>>>(W2, y1, st, y2, LCOLS);
  chan_stats_kernel<<<HCH, 256, 0, stream>>>(y2, LCOLS, st2);
  finalize2d_kernel<<<ELT, 256, 0, stream>>>(y2, st2, nullptr, x1CN, x1nc);

  // ---- stage 2 ----
  gemm_edge16_kernel<<<GEB, 256, 0, stream>>>(W3, x1nc, idxb, 2, y1, LCOLS);
  chan_stats_kernel<<<HCH, 256, 0, stream>>>(y1, LCOLS, st);
  gemm_normed_kernel<<<GEB, 256, 0, stream>>>(W4, y1, st, y2, LCOLS);
  chan_stats_kernel<<<HCH, 256, 0, stream>>>(y2, LCOLS, st2);
  finalize2d_kernel<<<ELT, 256, 0, stream>>>(y2, st2, x1CN, x2CN, x2nc);

  // ---- stage 3 ----
  gemm_edge16_kernel<<<GEB, 256, 0, stream>>>(W5, x2nc, idxb, 6, y1, LCOLS);
  chan_stats_kernel<<<HCH, 256, 0, stream>>>(y1, LCOLS, st);
  gemm_normed_kernel<<<GEB, 256, 0, stream>>>(W6, y1, st, y2, LCOLS);
  chan_stats_kernel<<<HCH, 256, 0, stream>>>(y2, LCOLS, st2);
  finalize2d_kernel<<<ELT, 256, 0, stream>>>(y2, st2, x2CN, x3CN, x3nc);

  // ---- stage 4 ---- (x4 f32[C,N] not needed: decoder uses g-bias + x4nc)
  gemm_edge16_kernel<<<GEB, 256, 0, stream>>>(W7, x3nc, idxb, 18, y1, LCOLS);
  chan_stats_kernel<<<HCH, 256, 0, stream>>>(y1, LCOLS, st);
  gemm_normed_kernel<<<GEB, 256, 0, stream>>>(W8, y1, st, y2, LCOLS);
  chan_stats_kernel<<<HCH, 256, 0, stream>>>(y2, LCOLS, st2);
  finalize2d_kernel<<<ELT, 256, 0, stream>>>(y2, st2, x3CN, x1CN /*scratch CN*/, x4nc);

  // ---- global feature g: W9 @ cat(x1..x4) ----
  Cat4 cg{{x1nc, x2nc, x3nc, x4nc}, {HCH, HCH, HCH, HCH}};
  gemm_cat_kernel<128, 256><<<NPTS / 16, 256, 0, stream>>>(W9, 256, cg, nullptr, ydec, NPTS);
  chan_stats_kernel<<<128, 256, 0, stream>>>(ydec, NPTS, st);
  rowmax_act_kernel<<<128, 256, 0, stream>>>(ydec, st, gvec, NPTS);
  bias10_kernel<<<1, 128, 0, stream>>>(W10, gvec, b10);

  // ---- decoder ----
  Cat4 c4{{x4nc, nullptr, nullptr, nullptr}, {HCH, 0, 0, 0}};
  gemm_cat_kernel<128, 64><<<NPTS / 16, 256, 0, stream>>>(W10 + 128, 192, c4, b10, ydec, NPTS);
  chan_stats_kernel<<<128, 256, 0, stream>>>(ydec, NPTS, st);
  norm_act_kernel<<<(128 * NPTS) / 256, 256, 0, stream>>>(ydec, st, d4nc, 128);

  Cat4 c3{{d4nc, x3nc, nullptr, nullptr}, {128, HCH, 0, 0}};
  gemm_cat_kernel<128, 192><<<NPTS / 16, 256, 0, stream>>>(W11, 192, c3, nullptr, ydec, NPTS);
  chan_stats_kernel<<<128, 256, 0, stream>>>(ydec, NPTS, st);
  norm_act_kernel<<<(128 * NPTS) / 256, 256, 0, stream>>>(ydec, st, d3nc, 128);

  Cat4 c2{{d3nc, x2nc, nullptr, nullptr}, {128, HCH, 0, 0}};
  gemm_cat_kernel<64, 192><<<NPTS / 32, 256, 0, stream>>>(W12, 192, c2, nullptr, ydec, NPTS);
  chan_stats_kernel<<<64, 256, 0, stream>>>(ydec, NPTS, st);
  norm_act_kernel<<<(64 * NPTS) / 256, 256, 0, stream>>>(ydec, st, d2nc, 64);

  Cat4 c1{{d2nc, x1nc, nullptr, nullptr}, {HCH, HCH, 0, 0}};
  gemm_cat_kernel<64, 128><<<NPTS / 32, 256, 0, stream>>>(W13, 128, c1, nullptr, ydec, NPTS);
  chan_stats_kernel<<<64, 256, 0, stream>>>(ydec, NPTS, st);
  norm_act_kernel<<<(64 * NPTS) / 256, 256, 0, stream>>>(ydec, st, d1nc, 64);

  // ---- output: per-object mean then Wr matvec ----
  obj_mean_kernel<<<2, 256, 0, stream>>>(d1nc, xm);
  final_out_kernel<<<4, 256, 0, stream>>>(Wr, xm, (float*)d_out);
}
